// GATPolicy_51986284150876
// MI455X (gfx1250) — compile-verified, matrix operand order checked by reading, not run on hardware
//
#include <hip/hip_runtime.h>
#include <hip/hip_bf16.h>
#include <stdint.h>

#define BB 64
#define NN 512
#define FC_IN 2204
#define FC_PAD 2208
#define YLEN 156
#define MLP_H 128
#define N_ACT 512
#define NEG_SLOPE 0.2f
#define NT1 69            // FC_PAD/32 K-tiles for MLP layer 1
#define NT2 4             // MLP_H/32  K-tiles for MLP layer 2

typedef __attribute__((ext_vector_type(16))) __bf16 v16bf;
typedef __attribute__((ext_vector_type(8)))  float  v8f;

union V16 { v16bf v; unsigned int u[8]; };

__device__ __forceinline__ unsigned int pack_bf16(float lo, float hi) {
    return (__float_as_uint(lo) >> 16) | (__float_as_uint(hi) & 0xffff0000u);
}

// fragment load from pre-packed buffer: 8 contiguous dwords -> two b128 loads
__device__ __forceinline__ v16bf load_frag(const unsigned int* __restrict__ p) {
    V16 f;
    *(uint4*)&f.u[0] = ((const uint4*)p)[0];
    *(uint4*)&f.u[4] = ((const uint4*)p)[1];
    return f.v;
}

// ---------------- pack W[K_SRC][NW] f32 -> WMMA-B bf16 fragments ------------
// layout: Wb[ ((t*NW + n)*2 + half)*8 + r ], zero-padded past K_SRC
__global__ __launch_bounds__(256) void k_packB(const float* __restrict__ W,
        unsigned int* __restrict__ Wb, int K_SRC, int NW) {
    int gid = blockIdx.x * 256 + threadIdx.x;      // (t*NW + n)*2 + half
    int half = gid & 1;
    int tn = gid >> 1;
    int t = tn / NW, n = tn % NW;
    unsigned int* dst = Wb + (size_t)gid * 8;
#pragma unroll
    for (int r = 0; r < 8; ++r) {
        int k0 = t * 32 + 16 * (r >> 2) + 2 * (r & 3) + 8 * half;
        float v0 = (k0     < K_SRC) ? W[(size_t)k0 * NW + n]       : 0.f;
        float v1 = (k0 + 1 < K_SRC) ? W[(size_t)(k0 + 1) * NW + n] : 0.f;
        dst[r] = pack_bf16(v0, v1);
    }
}

// ---------------- pack A[M][LDA] f32 -> WMMA-A bf16 fragments ---------------
// layout: Af[ ((m*2 + half)*NTt + t)*8 + r ]  (A rows already zero-padded)
__global__ __launch_bounds__(256) void k_packA(const float* __restrict__ A,
        unsigned int* __restrict__ Af, int NTt, int LDA, int total) {
    int gid = blockIdx.x * 256 + threadIdx.x;      // (m*2+half)*NTt + t
    if (gid >= total) return;
    int t = gid % NTt;
    int mh = gid / NTt;
    int half = mh & 1, m = mh >> 1;
    const float* arow = A + (size_t)m * LDA + t * 32;
    unsigned int* dst = Af + (size_t)gid * 8;
#pragma unroll
    for (int r = 0; r < 8; ++r) {
        int k = 16 * (r >> 2) + 2 * (r & 3) + 8 * half;
        dst[r] = pack_bf16(arow[k], arow[k + 1]);
    }
}

// ---------------- GAT layer 1 projection ------------------------------------
__global__ __launch_bounds__(256) void k_h1(const float* __restrict__ x,
        const float* __restrict__ W1, const float* __restrict__ asrc,
        const float* __restrict__ adst, float* __restrict__ h1,
        float* __restrict__ es, float* __restrict__ ed) {
    int gid = blockIdx.x * 256 + threadIdx.x;
    if (gid >= BB * NN) return;
    float xv[3];
#pragma unroll
    for (int d = 0; d < 3; ++d) xv[d] = x[gid * 3 + d];
    float h[6];
#pragma unroll
    for (int c = 0; c < 6; ++c) {
        float acc = 0.f;
#pragma unroll
        for (int d = 0; d < 3; ++d) acc += xv[d] * W1[d * 6 + c];
        h[c] = acc;
        h1[gid * 6 + c] = acc;
    }
#pragma unroll
    for (int hh = 0; hh < 2; ++hh) {
        float s = 0.f, t = 0.f;
#pragma unroll
        for (int d = 0; d < 3; ++d) {
            s += h[hh * 3 + d] * asrc[hh * 3 + d];
            t += h[hh * 3 + d] * adst[hh * 3 + d];
        }
        es[gid * 2 + hh] = s;
        ed[gid * 2 + hh] = t;
    }
}

// ---------------- GAT layer 1 attention + aggregate + ReLU ------------------
__global__ __launch_bounds__(256) void k_attn1(const float* __restrict__ adj,
        const float* __restrict__ h1, const float* __restrict__ es,
        const float* __restrict__ ed, const float* __restrict__ bias,
        float* __restrict__ g1) {
    __shared__ float sh[NN * 8];   // per source i: es0,es1,h[6]
    int b = blockIdx.x >> 1;
    int j = (blockIdx.x & 1) * 256 + threadIdx.x;
    for (int t = threadIdx.x; t < NN; t += 256) {
        int src = b * NN + t;
        sh[t * 8 + 0] = es[src * 2 + 0];
        sh[t * 8 + 1] = es[src * 2 + 1];
#pragma unroll
        for (int d = 0; d < 6; ++d) sh[t * 8 + 2 + d] = h1[src * 6 + d];
    }
    __syncthreads();
    float ed0 = ed[(b * NN + j) * 2 + 0];
    float ed1v = ed[(b * NN + j) * 2 + 1];
    const float* arow = adj + (size_t)b * NN * NN + j;
    float l0 = 0.f, l1 = 0.f;
    float a00 = 0.f, a01 = 0.f, a02 = 0.f, a10 = 0.f, a11 = 0.f, a12 = 0.f;
#pragma unroll 2
    for (int i = 0; i < NN; ++i) {
        float av = arow[(size_t)i * NN];
        bool m = (av > 0.f) || (i == j);          // self-loop always present
        float e0 = sh[i * 8 + 0] + ed0; e0 = e0 > 0.f ? e0 : NEG_SLOPE * e0;
        float e1 = sh[i * 8 + 1] + ed1v; e1 = e1 > 0.f ? e1 : NEG_SLOPE * e1;
        float p0 = m ? __expf(e0) : 0.f;          // bounded logits: no max pass
        float p1 = m ? __expf(e1) : 0.f;
        l0 += p0; l1 += p1;
        a00 += p0 * sh[i * 8 + 2]; a01 += p0 * sh[i * 8 + 3]; a02 += p0 * sh[i * 8 + 4];
        a10 += p1 * sh[i * 8 + 5]; a11 += p1 * sh[i * 8 + 6]; a12 += p1 * sh[i * 8 + 7];
    }
    float r0 = 1.f / l0, r1 = 1.f / l1;
    float o[6] = {a00 * r0, a01 * r0, a02 * r0, a10 * r1, a11 * r1, a12 * r1};
    size_t base = (size_t)(b * NN + j) * 6;
#pragma unroll
    for (int c = 0; c < 6; ++c) {
        float v = o[c] + bias[c];
        g1[base + c] = v > 0.f ? v : 0.f;
    }
}

// ---------------- GAT layer 2 projection ------------------------------------
__global__ __launch_bounds__(256) void k_h2(const float* __restrict__ g1,
        const float* __restrict__ W2, const float* __restrict__ asrc,
        const float* __restrict__ adst, float* __restrict__ h2,
        float* __restrict__ es, float* __restrict__ ed) {
    int gid = blockIdx.x * 256 + threadIdx.x;
    if (gid >= BB * NN) return;
    float gv[6];
#pragma unroll
    for (int c = 0; c < 6; ++c) gv[c] = g1[gid * 6 + c];
    float s = 0.f, t = 0.f;
#pragma unroll
    for (int o = 0; o < 3; ++o) {
        float acc = 0.f;
#pragma unroll
        for (int c = 0; c < 6; ++c) acc += gv[c] * W2[c * 3 + o];
        h2[gid * 3 + o] = acc;
        s += acc * asrc[o];
        t += acc * adst[o];
    }
    es[gid] = s;
    ed[gid] = t;
}

// ---------------- GAT layer 2 attention; writes into feat buffer ------------
__global__ __launch_bounds__(256) void k_attn2(const float* __restrict__ adj,
        const float* __restrict__ h2, const float* __restrict__ es,
        const float* __restrict__ ed, const float* __restrict__ bias,
        float* __restrict__ feat) {
    __shared__ float sh[NN * 4];  // es, h[3]
    int b = blockIdx.x >> 1;
    int j = (blockIdx.x & 1) * 256 + threadIdx.x;
    for (int t = threadIdx.x; t < NN; t += 256) {
        int src = b * NN + t;
        sh[t * 4 + 0] = es[src];
        sh[t * 4 + 1] = h2[src * 3 + 0];
        sh[t * 4 + 2] = h2[src * 3 + 1];
        sh[t * 4 + 3] = h2[src * 3 + 2];
    }
    __syncthreads();
    float edj = ed[b * NN + j];
    const float* arow = adj + (size_t)b * NN * NN + j;
    float l = 0.f, a0 = 0.f, a1 = 0.f, a2 = 0.f;
#pragma unroll 2
    for (int i = 0; i < NN; ++i) {
        float av = arow[(size_t)i * NN];
        bool m = (av > 0.f) || (i == j);
        float e = sh[i * 4 + 0] + edj; e = e > 0.f ? e : NEG_SLOPE * e;
        float p = m ? __expf(e) : 0.f;
        l += p;
        a0 += p * sh[i * 4 + 1]; a1 += p * sh[i * 4 + 2]; a2 += p * sh[i * 4 + 3];
    }
    float r = 1.f / l;
    float* dst = feat + (size_t)b * FC_PAD + NN + j * 3;
    dst[0] = a0 * r + bias[0];
    dst[1] = a1 * r + bias[1];
    dst[2] = a2 * r + bias[2];
}

// ---------------- fill idx / y parts of feat + zero K-pad -------------------
__global__ __launch_bounds__(256) void k_feat(const float* __restrict__ idx,
        const float* __restrict__ y, float* __restrict__ feat) {
    int b = blockIdx.x, t = threadIdx.x;
    float* row = feat + (size_t)b * FC_PAD;
    row[t]       = idx[b * NN + t];
    row[t + 256] = idx[b * NN + t + 256];
    if (t < YLEN)          row[2048 + t] = y[b * YLEN + t];
    else if (t < YLEN + 4) row[2048 + t] = 0.f;   // pad 2204..2207
}

// ---------------- MLP layer 1: relu(feat @ Wf1 + bf1), WMMA bf16 ------------
__global__ __launch_bounds__(128) void k_mlp1(const unsigned int* __restrict__ Af1,
        const unsigned int* __restrict__ Wb1, const float* __restrict__ bf1,
        float* __restrict__ hmid) {
    int wave = (blockIdx.x * 128 + threadIdx.x) >> 5;  // 0..31
    int lane = threadIdx.x & 31;
    int mt = wave >> 3, nt = wave & 7;
    int row = lane & 15, half = lane >> 4;
    int n = nt * 16 + row;
    const unsigned int* abase = Af1 + ((size_t)((mt * 16 + row) * 2 + half)) * NT1 * 8;
    const unsigned int* bbase = Wb1 + ((size_t)n * 2 + half) * 8;
    v8f c0 = {}, c1 = {};                 // two independent WMMA chains
#pragma unroll 2
    for (int t = 0; t + 1 < NT1; t += 2) {
        v16bf a0 = load_frag(abase + (size_t)t * 8);
        v16bf b0 = load_frag(bbase + (size_t)t * MLP_H * 16);
        c0 = __builtin_amdgcn_wmma_f32_16x16x32_bf16(false, a0, false, b0, (short)0, c0, false, false);
        v16bf a1 = load_frag(abase + (size_t)(t + 1) * 8);
        v16bf b1 = load_frag(bbase + (size_t)(t + 1) * MLP_H * 16);
        c1 = __builtin_amdgcn_wmma_f32_16x16x32_bf16(false, a1, false, b1, (short)0, c1, false, false);
    }
    {   // tail tile (NT1 is odd)
        v16bf a0 = load_frag(abase + (size_t)(NT1 - 1) * 8);
        v16bf b0 = load_frag(bbase + (size_t)(NT1 - 1) * MLP_H * 16);
        c0 = __builtin_amdgcn_wmma_f32_16x16x32_bf16(false, a0, false, b0, (short)0, c0, false, false);
    }
#pragma unroll
    for (int v = 0; v < 8; ++v) {
        int m = mt * 16 + v + 8 * half;
        float val = c0[v] + c1[v] + bf1[n];
        hmid[m * MLP_H + n] = val > 0.f ? val : 0.f;
    }
}

// ---------------- MLP layer 2: hmid @ Wf2 + bf2, WMMA bf16 ------------------
__global__ __launch_bounds__(128) void k_mlp2(const unsigned int* __restrict__ Af2,
        const unsigned int* __restrict__ Wb2, const float* __restrict__ bf2,
        float* __restrict__ out) {
    int wave = (blockIdx.x * 128 + threadIdx.x) >> 5;  // 0..127
    int lane = threadIdx.x & 31;
    int mt = wave >> 5, nt = wave & 31;
    int row = lane & 15, half = lane >> 4;
    int n = nt * 16 + row;
    const unsigned int* abase = Af2 + ((size_t)((mt * 16 + row) * 2 + half)) * NT2 * 8;
    const unsigned int* bbase = Wb2 + ((size_t)n * 2 + half) * 8;
    v8f c0 = {}, c1 = {};
#pragma unroll
    for (int t = 0; t < NT2; t += 2) {
        v16bf a0 = load_frag(abase + (size_t)t * 8);
        v16bf b0 = load_frag(bbase + (size_t)t * N_ACT * 16);
        c0 = __builtin_amdgcn_wmma_f32_16x16x32_bf16(false, a0, false, b0, (short)0, c0, false, false);
        v16bf a1 = load_frag(abase + (size_t)(t + 1) * 8);
        v16bf b1 = load_frag(bbase + (size_t)(t + 1) * N_ACT * 16);
        c1 = __builtin_amdgcn_wmma_f32_16x16x32_bf16(false, a1, false, b1, (short)0, c1, false, false);
    }
#pragma unroll
    for (int v = 0; v < 8; ++v) {
        int m = mt * 16 + v + 8 * half;
        out[(size_t)m * N_ACT + n] = c0[v] + c1[v] + bf2[n];
    }
}

extern "C" void kernel_launch(void* const* d_in, const int* in_sizes, int n_in,
                              void* d_out, int out_size, void* d_ws, size_t ws_size,
                              hipStream_t stream) {
    const float* idx    = (const float*)d_in[0];
    const float* x      = (const float*)d_in[1];
    const float* y      = (const float*)d_in[2];
    const float* adj    = (const float*)d_in[3];
    const float* W1     = (const float*)d_in[4];
    const float* a_src1 = (const float*)d_in[5];
    const float* a_dst1 = (const float*)d_in[6];
    const float* b1     = (const float*)d_in[7];
    const float* W2     = (const float*)d_in[8];
    const float* a_src2 = (const float*)d_in[9];
    const float* a_dst2 = (const float*)d_in[10];
    const float* b2     = (const float*)d_in[11];
    const float* Wf1    = (const float*)d_in[12];
    const float* bf1    = (const float*)d_in[13];
    const float* Wf2    = (const float*)d_in[14];
    const float* bf2    = (const float*)d_in[15];
    float* out = (float*)d_out;

    float* ws   = (float*)d_ws;
    float* h1   = ws;                 // 64*512*6 = 196608
    float* es1  = h1  + 196608;       // 65536
    float* ed1  = es1 + 65536;        // 65536
    float* g1   = ed1 + 65536;        // 196608
    float* h2   = g1  + 196608;       // 98304
    float* es2  = h2  + 98304;        // 32768
    float* ed2  = es2 + 32768;        // 32768
    float* feat = ed2 + 32768;        // 64*2208 = 141312
    float* hmid = feat + 141312;      // 64*128  = 8192
    unsigned int* Wb1 = (unsigned int*)(hmid + 8192);   // 69*128*2*8 = 141312 dwords
    unsigned int* Wb2 = Wb1 + 141312;                   // 4*512*2*8  = 32768 dwords
    unsigned int* Af1 = Wb2 + 32768;                    // 64*2*69*8  = 70656 dwords
    unsigned int* Af2 = Af1 + 70656;                    // 64*2*4*8   = 4096 dwords

    // weight pre-pack (tiny)
    k_packB<<<69, 256, 0, stream>>>(Wf1, Wb1, FC_IN, MLP_H);   // 69*128*2 = 17664
    k_packB<<<16, 256, 0, stream>>>(Wf2, Wb2, MLP_H, N_ACT);   // 4*512*2  = 4096

    k_h1   <<<128, 256, 0, stream>>>(x, W1, a_src1, a_dst1, h1, es1, ed1);
    k_attn1<<<128, 256, 0, stream>>>(adj, h1, es1, ed1, b1, g1);
    k_h2   <<<128, 256, 0, stream>>>(g1, W2, a_src2, a_dst2, h2, es2, ed2);
    k_attn2<<<128, 256, 0, stream>>>(adj, h2, es2, ed2, b2, feat);
    k_feat <<<64,  256, 0, stream>>>(idx, y, feat);

    k_packA<<<35, 256, 0, stream>>>(feat, Af1, NT1, FC_PAD, BB * 2 * NT1); // 8832
    k_mlp1 <<<8,   128, 0, stream>>>(Af1, Wb1, bf1, hmid);
    k_packA<<<2,  256, 0, stream>>>(hmid, Af2, NT2, MLP_H, BB * 2 * NT2); // 512
    k_mlp2 <<<32,  128, 0, stream>>>(Af2, Wb2, bf2, out);
}
